// PMGCN_48988396978448
// MI455X (gfx1250) — matrix operand order repaired and checked by reading, not compiled, not run on hardware
//
#include <hip/hip_runtime.h>
#include <hip/hip_bf16.h>

// ---------------------------------------------------------------------------
// GATv2 GNN forward for MI455X (gfx1250, wave32).
// Node transforms via v_wmma_f32_16x16x32_f16 (f16 in, f32 acc); weight
// matrices are pre-swizzled into WMMA B-fragment order so each fragment is a
// single contiguous 32B load per lane. Edge phase = 3 streamed passes
// (logit+segmax, exp+segsum, weighted scatter), wave-per-edge with coalesced
// float2 lanes; HBM-bound by design (~3.5 GB @ 23.3 TB/s).
// ---------------------------------------------------------------------------

typedef __attribute__((ext_vector_type(16))) _Float16 v16h;
typedef __attribute__((ext_vector_type(8)))  float    v8f;

#define NNODES 65536
#define NEDGE  1048576
#define NEDGE2 (NEDGE + NNODES)   // + self loops
#define NGRAPH 256
#define HIDW   64

// ---- monotonic float <-> uint map for atomic max --------------------------
__device__ __forceinline__ unsigned fflip(float f) {
    unsigned b = __float_as_uint(f);
    return (b & 0x80000000u) ? ~b : (b | 0x80000000u);
}
__device__ __forceinline__ float funflip(unsigned u) {
    unsigned b = (u & 0x80000000u) ? (u & 0x7fffffffu) : ~u;
    return __uint_as_float(b);
}

// ---- f32 weights -> f16, pre-swizzled into WMMA B-fragment layout ---------
// Fragment order: frag = kc*4 + nt (kc = K-chunk of 32, nt = N-tile of 16).
// Within a fragment, lane L holds 16 contiguous halfs:
//   element e = W[kc*32 + (e<8 ? e : e+8) + (L>>4)*8][nt*16 + (L&15)]
__global__ void k_swizzle_w(const float* __restrict__ src,
                            _Float16* __restrict__ dst) {
    int i = blockIdx.x * blockDim.x + threadIdx.x;   // 0..4095
    if (i >= 4096) return;
    int e    = i & 15;
    int lane = (i >> 4) & 31;
    int frag = i >> 9;           // 0..7
    int kc   = frag >> 2;
    int nt   = frag & 3;
    int k = kc * 32 + (e < 8 ? e : e + 8) + (lane >> 4) * 8;
    int n = nt * 16 + (lane & 15);
    dst[i] = (_Float16)src[k * HIDW + n];
}

// ---- WMMA GEMM: Y[rows,64] = X[rows,64] @ W[64,64] + b --------------------
// One wave -> 16 rows x 64 cols. mode: 0 identity row map,
// 1 type-node scatter (g*256 + p), 2 attr-node scatter (g*256 + 128 + p).
__global__ void k_gemm64(const float* __restrict__ X,
                         const _Float16* __restrict__ Wsw,
                         const float* __restrict__ bias,
                         float* __restrict__ Y,
                         int ntiles, int mode) {
    const int lane = threadIdx.x & 31;
    const int wave = threadIdx.x >> 5;
    const int tile = blockIdx.x * (blockDim.x >> 5) + wave;
    if (tile >= ntiles) return;              // wave-uniform: EXEC stays all-1s

    const int row0 = tile << 4;
    const int m    = lane & 15;              // A-matrix row within tile
    const int hsel = lane >> 4;              // lane-half selector
    const int n    = lane & 15;              // B/C column within n-tile

    // A fragments (16x32 f16, ISA 7.12.2 layout), two K-chunks
    const float* xrow = X + (size_t)(row0 + m) * HIDW;
    v16h a0, a1;
#pragma unroll
    for (int e = 0; e < 16; ++e) {
        int k = (e < 8 ? e : e + 8) + hsel * 8;   // K within 32
        a0[e] = (_Float16)xrow[k];
        a1[e] = (_Float16)xrow[32 + k];
    }

    // B fragments: pre-swizzled, one 32B contiguous vector load per fragment
    const v16h* Bw = (const v16h*)Wsw;
    v16h b0[4], b1[4];
#pragma unroll
    for (int nt = 0; nt < 4; ++nt) {
        b0[nt] = Bw[(0 * 4 + nt) * 32 + lane];
        b1[nt] = Bw[(1 * 4 + nt) * 32 + lane];
    }

    // Accumulators pre-loaded with bias (C/D layout: M = r + 8*hsel, N = n)
    v8f acc[4];
#pragma unroll
    for (int nt = 0; nt < 4; ++nt) {
        float bv = bias[nt * 16 + n];
#pragma unroll
        for (int r = 0; r < 8; ++r) acc[nt][r] = bv;
    }

#pragma unroll
    for (int nt = 0; nt < 4; ++nt) {
        acc[nt] = __builtin_amdgcn_wmma_f32_16x16x32_f16(
            false, a0, false, b0[nt], (short)0, acc[nt], false, false);
        acc[nt] = __builtin_amdgcn_wmma_f32_16x16x32_f16(
            false, a1, false, b1[nt], (short)0, acc[nt], false, false);
    }

#pragma unroll
    for (int r = 0; r < 8; ++r) {
        int inrow = row0 + r + hsel * 8;
        int orow;
        if (mode == 0) orow = inrow;
        else {
            int g = inrow >> 7, p = inrow & 127;
            orow = g * 256 + p + (mode == 2 ? 128 : 0);
        }
#pragma unroll
        for (int nt = 0; nt < 4; ++nt)
            Y[(size_t)orow * HIDW + nt * 16 + n] = acc[nt][r];
    }
}

// ---- per-layer init: zero accumulator, reset segment max/sum --------------
__global__ void k_init(float* __restrict__ out, unsigned* __restrict__ segmax,
                       float* __restrict__ segsum) {
    int i = blockIdx.x * blockDim.x + threadIdx.x;
    if (i < NNODES * HIDW) out[i] = 0.f;
    if (i < NNODES) { segmax[i] = 0u; segsum[i] = 0.f; }
}

// ---- edge pass 1: logit = att . leaky(xl[s]+xr[d], 0.2); segmax -----------
__global__ void k_edge_logit(const int* __restrict__ ei,
                             const float* __restrict__ xl,
                             const float* __restrict__ xr,
                             const float* __restrict__ att,
                             float* __restrict__ elogit,
                             unsigned* __restrict__ segmax) {
    long long t = (long long)blockIdx.x * blockDim.x + threadIdx.x;
    int e = (int)(t >> 5);
    int lane = threadIdx.x & 31;
    if (e >= NEDGE2) return;
    int s, d;
    if (e < NEDGE) { s = ei[e]; d = ei[NEDGE + e]; }
    else           { s = d = e - NEDGE; }

    float2 l = ((const float2*)(xl + (size_t)s * HIDW))[lane];
    float2 r = ((const float2*)(xr + (size_t)d * HIDW))[lane];
    float2 a = ((const float2*)att)[lane];
    float e0 = l.x + r.x; e0 = e0 > 0.f ? e0 : 0.2f * e0;
    float e1 = l.y + r.y; e1 = e1 > 0.f ? e1 : 0.2f * e1;
    float part = a.x * e0 + a.y * e1;
#pragma unroll
    for (int m = 16; m; m >>= 1) part += __shfl_xor(part, m, 32);
    if (lane == 0) {
        elogit[e] = part;
        atomicMax(segmax + d, fflip(part));
    }
}

// ---- edge pass 2: p = exp(logit - max[d]); segsum -------------------------
__global__ void k_edge_expsum(const int* __restrict__ ei,
                              float* __restrict__ elogit,
                              const unsigned* __restrict__ segmax,
                              float* __restrict__ segsum) {
    int e = blockIdx.x * blockDim.x + threadIdx.x;
    if (e >= NEDGE2) return;
    int d = (e < NEDGE) ? ei[NEDGE + e] : e - NEDGE;
    float p = __expf(elogit[e] - funflip(segmax[d]));
    elogit[e] = p;
    atomicAdd(segsum + d, p);
}

// ---- edge pass 3: out[d] += alpha * xl[s] ---------------------------------
__global__ void k_edge_scatter(const int* __restrict__ ei,
                               const float* __restrict__ elogit,
                               const float* __restrict__ segsum,
                               const float* __restrict__ xl,
                               float* __restrict__ out) {
    long long t = (long long)blockIdx.x * blockDim.x + threadIdx.x;
    int e = (int)(t >> 5);
    int lane = threadIdx.x & 31;
    if (e >= NEDGE2) return;
    int s, d;
    if (e < NEDGE) { s = ei[e]; d = ei[NEDGE + e]; }
    else           { s = d = e - NEDGE; }
    float alpha = elogit[e] / segsum[d];
    float2 l = ((const float2*)(xl + (size_t)s * HIDW))[lane];
    atomicAdd(out + (size_t)d * HIDW + lane * 2,     l.x * alpha);
    atomicAdd(out + (size_t)d * HIDW + lane * 2 + 1, l.y * alpha);
}

// ---- bias + leaky_relu(0.01) in place -------------------------------------
__global__ void k_bias_lrelu(float* __restrict__ x,
                             const float* __restrict__ bias) {
    int i = blockIdx.x * blockDim.x + threadIdx.x;
    if (i >= NNODES * HIDW) return;
    float v = x[i] + bias[i & 63];
    x[i] = v > 0.f ? v : 0.01f * v;
}

// ---- mean pool over exactly 256 nodes per graph ---------------------------
__global__ void k_pool(const float* __restrict__ x, float* __restrict__ pooled) {
    int g = blockIdx.x, h = threadIdx.x;
    float s = 0.f;
    for (int p = 0; p < 256; ++p)
        s += x[((size_t)g * 256 + p) * HIDW + h];
    pooled[g * HIDW + h] = s * (1.f / 256.f);
}

// ---- global features embed: [256,16] @ [16,64] + b ------------------------
__global__ void k_glob(const float* __restrict__ gf, const float* __restrict__ W,
                       const float* __restrict__ b, float* __restrict__ ge) {
    int g = blockIdx.x, o = threadIdx.x;
    float s = b[o];
    for (int k = 0; k < 16; ++k) s += gf[g * 16 + k] * W[k * HIDW + o];
    ge[g * HIDW + o] = s;
}

// ---- final linear: concat(pooled, ge) @ lin_W[128,64] + b -----------------
__global__ void k_final_lin(const float* __restrict__ pooled,
                            const float* __restrict__ ge,
                            const float* __restrict__ W,
                            const float* __restrict__ b,
                            float* __restrict__ out_emb) {
    int g = blockIdx.x, o = threadIdx.x;
    float s = b[o];
    for (int k = 0; k < 64; ++k) s += pooled[g * HIDW + k] * W[k * HIDW + o];
    for (int k = 0; k < 64; ++k) s += ge[g * HIDW + k] * W[(64 + k) * HIDW + o];
    out_emb[g * HIDW + o] = s;
}

// ---- softmax head: softmax(emb @ sm_W[64,64] + b) -------------------------
__global__ void k_softmax(const float* __restrict__ emb,
                          const float* __restrict__ W,
                          const float* __restrict__ b,
                          float* __restrict__ prob) {
    __shared__ float sh[64];
    int g = blockIdx.x, t = threadIdx.x;
    float s = b[t];
    for (int k = 0; k < 64; ++k) s += emb[g * HIDW + k] * W[k * 64 + t];
    sh[t] = s; __syncthreads();
    float m = sh[0];
    for (int k = 1; k < 64; ++k) m = fmaxf(m, sh[k]);
    float ex = __expf(s - m);
    __syncthreads(); sh[t] = ex; __syncthreads();
    float sum = 0.f;
    for (int k = 0; k < 64; ++k) sum += sh[k];
    prob[g * 64 + t] = ex / sum;
}

extern "C" void kernel_launch(void* const* d_in, const int* in_sizes, int n_in,
                              void* d_out, int out_size, void* d_ws, size_t ws_size,
                              hipStream_t stream) {
    (void)in_sizes; (void)n_in; (void)out_size; (void)ws_size;

    const float* type_nodes = (const float*)d_in[0];
    const float* attr_nodes = (const float*)d_in[1];
    const int*   ei         = (const int*)d_in[2];
    const float* gfeat      = (const float*)d_in[5];
    // params (dict order, flattened)
    const float* type_W = (const float*)d_in[7];
    const float* type_b = (const float*)d_in[8];
    const float* attr_W = (const float*)d_in[9];
    const float* attr_b = (const float*)d_in[10];
    const float* glob_W = (const float*)d_in[11];
    const float* glob_b = (const float*)d_in[12];
    const float* lin_W  = (const float*)d_in[13];
    const float* lin_b  = (const float*)d_in[14];
    const float* sm_W   = (const float*)d_in[15];
    const float* sm_b   = (const float*)d_in[16];

    // ---- workspace carve-up (~55.5 MB) ----
    float*    xbuf   = (float*)d_ws;                    // N*64 (x / accum)
    float*    xl     = xbuf + (size_t)NNODES * HIDW;    // N*64
    float*    xr     = xl   + (size_t)NNODES * HIDW;    // N*64
    float*    elogit = xr   + (size_t)NNODES * HIDW;    // E2
    unsigned* segmax = (unsigned*)(elogit + NEDGE2);    // N
    float*    segsum = (float*)(segmax + NNODES);       // N
    float*    pooled = segsum + NNODES;                 // 256*64
    float*    gemb   = pooled + NGRAPH * HIDW;          // 256*64
    _Float16* wh     = (_Float16*)(gemb + NGRAPH * HIDW); // 8 * 64*64 f16, swizzled

    // ---- convert + swizzle 8 weight matrices to f16 B-fragment order ----
    const float* wsrc[8] = {
        type_W, attr_W,
        (const float*)d_in[17], (const float*)d_in[19],   // c0 Wl, Wr
        (const float*)d_in[23], (const float*)d_in[25],   // c1
        (const float*)d_in[29], (const float*)d_in[31] }; // c2
    for (int w = 0; w < 8; ++w)
        k_swizzle_w<<<16, 256, 0, stream>>>(wsrc[w], wh + (size_t)w * 4096);

    // ---- embedding GEMMs with fused gather_id scatter ----
    // 32768 rows -> 2048 tiles -> 512 blocks of 4 waves
    k_gemm64<<<512, 128, 0, stream>>>(type_nodes, wh + 0 * 4096, type_b, xbuf, 2048, 1);
    k_gemm64<<<512, 128, 0, stream>>>(attr_nodes, wh + 1 * 4096, attr_b, xbuf, 2048, 2);

    const int EB = (NEDGE2 * 32) / 256;      // wave-per-edge grids
    // ---- 3 GATv2 layers ----
    for (int l = 0; l < 3; ++l) {
        const float* bl   = (const float*)d_in[18 + 6 * l];
        const float* br   = (const float*)d_in[20 + 6 * l];
        const float* att  = (const float*)d_in[21 + 6 * l];
        const float* bias = (const float*)d_in[22 + 6 * l];
        _Float16* whl = wh + (size_t)(2 + 2 * l) * 4096;
        _Float16* whr = wh + (size_t)(3 + 2 * l) * 4096;

        // 65536 rows -> 4096 tiles -> 1024 blocks
        k_gemm64<<<1024, 128, 0, stream>>>(xbuf, whl, bl, xl, 4096, 0);
        k_gemm64<<<1024, 128, 0, stream>>>(xbuf, whr, br, xr, 4096, 0);

        k_init<<<(NNODES * HIDW) / 256, 256, 0, stream>>>(xbuf, segmax, segsum);
        k_edge_logit<<<EB, 256, 0, stream>>>(ei, xl, xr, att, elogit, segmax);
        k_edge_expsum<<<NEDGE2 / 256, 256, 0, stream>>>(ei, elogit, segmax, segsum);
        k_edge_scatter<<<EB, 256, 0, stream>>>(ei, elogit, segsum, xl, xbuf);
        k_bias_lrelu<<<(NNODES * HIDW) / 256, 256, 0, stream>>>(xbuf, bias);
    }

    // ---- readout ----
    float* out_prob = (float*)d_out;                    // [256, 64]
    float* out_emb  = (float*)d_out + NGRAPH * HIDW;    // [256, 64]
    k_pool<<<NGRAPH, 64, 0, stream>>>(xbuf, pooled);
    k_glob<<<NGRAPH, 64, 0, stream>>>(gfeat, glob_W, glob_b, gemb);
    k_final_lin<<<NGRAPH, 64, 0, stream>>>(pooled, gemb, lin_W, lin_b, out_emb);
    k_softmax<<<NGRAPH, 64, 0, stream>>>(out_emb, sm_W, sm_b, out_prob);
}
